// EnTransformerBlock_26706106646672
// MI455X (gfx1250) — matrix operand order repaired and checked by reading, not compiled, not run on hardware
//
#include <hip/hip_runtime.h>

typedef __attribute__((ext_vector_type(16))) __bf16 v16bf;
typedef __attribute__((ext_vector_type(8)))  float  v8f;

#define WMMA_BF16(A_, B_, C_) \
  __builtin_amdgcn_wmma_f32_16x16x32_bf16(false, (A_), false, (B_), (short)0, (C_), false, false)

// Problem constants
#define NN   256
#define FF   128
#define HIDN 64
#define PACK_FRAGS 40   // 32 frags We1_rad + 8 frags We2
#define PACK_ELEMS (PACK_FRAGS * 512)

__device__ __forceinline__ float selu_f(float x) {
  // scale = 1.0507009873554805, scale*alpha = 1.7580993408473766
  return x > 0.f ? 1.0507009873554805f * x
                 : 1.7580993408473766f * (__expf(x) - 1.f);
}

__device__ __forceinline__ void wait_asynccnt0() {
#if __has_builtin(__builtin_amdgcn_s_wait_asynccnt)
  __builtin_amdgcn_s_wait_asynccnt(0);
#else
  asm volatile("s_wait_asynccnt 0x0" ::: "memory");
#endif
}

// ---------------------------------------------------------------------------
// Kernel 0: pre-pack We1_rad (256x64) + We2 (64x64) into bf16 WMMA B-fragment
// order, so every receiver block can async-DMA the identical 40KB blob to LDS.
//   B layout (32xK-block): lane 0-15: N=lane,   elem e -> K = 32t + 16*(e>>3) + (e&7)
//                          lane16-31: N=lane-16,elem e -> K = 32t + 8 + 16*(e>>3) + (e&7)
// ---------------------------------------------------------------------------
__global__ __launch_bounds__(256) void pack_weights_kernel(
    const float* __restrict__ We1, const float* __restrict__ We2,
    __bf16* __restrict__ pack) {
  for (int idx = threadIdx.x; idx < PACK_ELEMS; idx += 256) {
    const int frag = idx >> 9, rem = idx & 511, ln = rem >> 4, e = rem & 15;
    float v;
    if (frag < 32) {
      const int t = frag >> 2, j = frag & 3;
      const int k = 32 * t + 8 * (ln >> 4) + 16 * (e >> 3) + (e & 7);
      const int col = 16 * j + (ln & 15);
      v = We1[(256 + k) * 64 + col];
    } else {
      const int f2 = frag - 32;
      const int t = f2 >> 2, j = f2 & 3;
      const int k = 32 * t + 8 * (ln >> 4) + 16 * (e >> 3) + (e & 7);
      const int col = 16 * j + (ln & 15);
      v = We2[k * 64 + col];
    }
    pack[idx] = (__bf16)v;
  }
}

// ---------------------------------------------------------------------------
// Kernel 1: per-node precompute
//   Fs'[n] = feat[n] @ We1[0:128]           (sender block of edge MLP layer1)
//   Fr'[n] = feat[n] @ We1[128:256] + be1   (receiver block + bias folded)
//   q[n]   = feat[n] @ W_q ;  v[n] = feat[n] @ W_v
// ---------------------------------------------------------------------------
__global__ __launch_bounds__(128) void node_pre_kernel(
    const float* __restrict__ feat, const float* __restrict__ We1,
    const float* __restrict__ be1, const float* __restrict__ W_q,
    const float* __restrict__ W_v,
    float* __restrict__ wsFs, float* __restrict__ wsFr,
    float* __restrict__ wsQ, float* __restrict__ wsV) {
  __shared__ float f[FF];
  const int n = blockIdx.x, t = threadIdx.x;
  f[t] = feat[n * FF + t];
  __syncthreads();
  float q = 0.f, v = 0.f;
#pragma unroll 8
  for (int d = 0; d < FF; ++d) {
    q = fmaf(f[d], W_q[d * FF + t], q);
    v = fmaf(f[d], W_v[d * FF + t], v);
  }
  wsQ[n * FF + t] = q;
  wsV[n * FF + t] = v;
  if (t < HIDN) {
    float a = 0.f, b = 0.f;
#pragma unroll 8
    for (int d = 0; d < FF; ++d) {
      a = fmaf(f[d], We1[d * HIDN + t], a);
      b = fmaf(f[d], We1[(FF + d) * HIDN + t], b);
    }
    wsFs[n * HIDN + t] = a;
    wsFr[n * HIDN + t] = b + be1[t];
  }
}

// ---------------------------------------------------------------------------
// Kernel 2: per-receiver fused edge pipeline (1 block per receiver r)
//   async-DMA weight fragments -> radial bf16 in A-frag registers
//   -> WMMA GEMM1 (K=256) -> selu -> WMMA GEMM2 (64x64) -> selu
//   -> WMMA GEMM3 ([Wq_r | W_vec], 64x40) -> segment softmax -> aggregation
// ---------------------------------------------------------------------------
__global__ __launch_bounds__(256) void edge_attn_kernel(
    const float* __restrict__ pos,
    const float* __restrict__ be2, const float* __restrict__ W_k,
    const float* __restrict__ W_vec, const __bf16* __restrict__ wsPack,
    const float* __restrict__ wsFs, const float* __restrict__ wsFr,
    const float* __restrict__ wsQ, const float* __restrict__ wsV,
    float* __restrict__ wsInv, float* __restrict__ wsVec) {
  __shared__ __attribute__((aligned(32))) __bf16 sBpack[PACK_ELEMS]; // We1_rad + We2 frags
  __shared__ __attribute__((aligned(32))) __bf16 sB3[6][32][16];     // [Wq_r|W_vec] 64x48
  __shared__ float  sWq[64][8];
  __shared__ float  sFr[64];
  __shared__ float  sBe2[64];
  __shared__ float  sPosR[96];
  __shared__ __attribute__((aligned(32))) __bf16 sH[8][16][64];      // per-wave h staging
  __shared__ float  sLogit[256][8];    // logits, then alpha in-place
  __shared__ float  sCoeff[256][32];
  __shared__ float  sPInv[8][128];     // deterministic per-wave partials
  __shared__ float  sPVec[8][96];

  const int tid  = threadIdx.x;
  const int r    = blockIdx.x;
  const int w    = tid >> 5;
  const int lane = tid & 31;
  const int mrow = lane & 15;
  const int hi   = lane >> 4;

  // ---- async DMA of the 40KB weight-fragment blob into LDS (ASYNCcnt) ----
  {
    const unsigned ldsBase =
        (unsigned)(unsigned long long)(const void*)&sBpack[0];
    const unsigned long long gbase = (unsigned long long)(const void*)wsPack;
#pragma unroll
    for (int i = 0; i < 10; ++i) {           // 2560 x b128 chunks / 256 threads
      const unsigned byteOff = (unsigned)(tid + 256 * i) * 16u;
      const unsigned ldsAddr = ldsBase + byteOff;
      const unsigned long long ga = gbase + (unsigned long long)byteOff;
      asm volatile("global_load_async_to_lds_b128 %0, %1, off"
                   :: "v"(ldsAddr), "v"(ga) : "memory");
    }
  }

  // ---- overlapped per-receiver setup while DMA is in flight ----
  if (tid < 64) { sFr[tid] = wsFr[r * 64 + tid]; sBe2[tid] = be2[tid]; }
  if (tid < 96) sPosR[tid] = pos[r * 96 + tid];

  // Wq_r[k][h] = sum_d W_k[k][16h+d] * q[r][16h+d]   (folds W_k GEMM + dot)
  for (int idx = tid; idx < 512; idx += 256) {
    const int k = idx >> 3, h = idx & 7;
    const float* wk = W_k + k * 128 + h * 16;
    const float* qr = wsQ + r * 128 + h * 16;
    float a = 0.f;
#pragma unroll
    for (int d = 0; d < 16; ++d) a = fmaf(wk[d], qr[d], a);
    sWq[k][h] = a;
  }
  __syncthreads();

  for (int fid = tid; fid < 6 * 512; fid += 256) {
    const int frag = fid >> 9, rem = fid & 511, ln = rem >> 4, e = rem & 15;
    const int t = frag / 3, j = frag % 3;
    const int k = 32 * t + 8 * (ln >> 4) + 16 * (e >> 3) + (e & 7);
    const int col = 16 * j + (ln & 15);
    float vsrc;
    if (col < 8)       vsrc = sWq[k][col];
    else if (col < 40) vsrc = W_vec[k * 32 + (col - 8)];
    else               vsrc = 0.f;
    sB3[frag][ln][e] = (__bf16)vsrc;
  }
  wait_asynccnt0();
  __syncthreads();

  // ---- per-sender-tile pipeline: 16 chunks of 16 senders, 2 per wave ----
  for (int it = 0; it < 2; ++it) {
    const int chunk = w + 8 * it;
    const int sbase = chunk * 16;
    const int srow  = sbase + mrow;   // this lane's A-row sender

    // Radial Bessel*envelope computed directly into A-fragment registers.
    // 16-bit A layout: lane row M = lane&15; elem e -> K = 32t + 8*(2*(e>=8)+hi) + (e&7)
    // With K = 8*v + b this means: lanes<16 handle even v, lanes>=16 odd v.
    v16bf A[8];
#pragma unroll
    for (int t = 0; t < 8; ++t) {
#pragma unroll
      for (int half = 0; half < 2; ++half) {
        const int vch = 4 * t + 2 * half + hi;
        const float dx = pos[(srow * 32 + vch) * 3 + 0] - sPosR[vch * 3 + 0];
        const float dy = pos[(srow * 32 + vch) * 3 + 1] - sPosR[vch * 3 + 1];
        const float dz = pos[(srow * 32 + vch) * 3 + 2] - sPosR[vch * 3 + 2];
        const float len =
            __builtin_amdgcn_sqrtf(fmaf(dx, dx, fmaf(dy, dy, fmaf(dz, dz, 1e-18f))));
        const float x  = fmaxf(len, 1e-6f);
        const float th = 0.6283185307179586f * x;          // pi*x/5
        const float s1 = __sinf(th);
        const float twoc = 2.f * __cosf(th);
        const float te = 2.f * (1.f - 0.2f * len);
        const float env =
            (te > 0.f) ? 1.2f * __expf(0.5f - __builtin_amdgcn_rcpf(te)) : 0.f;
        const float scale =
            0.6324555320336759f * env * __builtin_amdgcn_rcpf(x); // sqrt(2/5)*env/x
        float bk = s1, bkm1 = 0.f;                         // sin(k*th) recurrence
#pragma unroll
        for (int b = 0; b < 8; ++b) {
          A[t][half * 8 + b] = (__bf16)(bk * scale);
          const float nx = fmaf(twoc, bk, -bkm1);
          bkm1 = bk; bk = nx;
        }
      }
    }

    // GEMM1: radial[16x256] @ We1_rad[256x64]; h1 = selu(acc + Fs[s] + Fr'[r])
#pragma unroll
    for (int j = 0; j < 4; ++j) {
      v8f c = {};
#pragma unroll
      for (int t = 0; t < 8; ++t) {
        const v16bf b = *(const v16bf*)&sBpack[((t * 4 + j) * 32 + lane) * 16];
        c = WMMA_BF16(A[t], b, c);
      }
      const int col = 16 * j + mrow;   // C layout: lane -> column
      const float fr = sFr[col];
#pragma unroll
      for (int i = 0; i < 8; ++i) {    // VGPR i -> row i + 8*hi
        const int row = i + 8 * hi;
        const float fs = wsFs[(sbase + row) * 64 + col];
        sH[w][row][col] = (__bf16)selu_f(c[i] + fs + fr);
      }
    }
    __syncthreads();

    // GEMM2: h1[16x64] @ We2[64x64]; h2 = selu(acc + be2)
    v16bf Ah[2];
#pragma unroll
    for (int t = 0; t < 2; ++t)
#pragma unroll
      for (int e = 0; e < 16; ++e)
        Ah[t][e] = sH[w][mrow][32 * t + 8 * (((e >> 3) << 1) + hi) + (e & 7)];
    v8f c2[4];
#pragma unroll
    for (int j = 0; j < 4; ++j) {
      v8f c = {};
      c = WMMA_BF16(Ah[0], *(const v16bf*)&sBpack[((32 + j) * 32 + lane) * 16], c);
      c = WMMA_BF16(Ah[1], *(const v16bf*)&sBpack[((36 + j) * 32 + lane) * 16], c);
      c2[j] = c;
    }
    __syncthreads();
#pragma unroll
    for (int j = 0; j < 4; ++j) {
      const int col = 16 * j + mrow;
      const float b2 = sBe2[col];
#pragma unroll
      for (int i = 0; i < 8; ++i)
        sH[w][i + 8 * hi][col] = (__bf16)selu_f(c2[j][i] + b2);
    }
    __syncthreads();

    // GEMM3: h2[16x64] @ [Wq_r | W_vec][64x48] -> logits (cols 0..7) + coeff (8..39)
#pragma unroll
    for (int t = 0; t < 2; ++t)
#pragma unroll
      for (int e = 0; e < 16; ++e)
        Ah[t][e] = sH[w][mrow][32 * t + 8 * (((e >> 3) << 1) + hi) + (e & 7)];
#pragma unroll
    for (int j = 0; j < 3; ++j) {
      v8f c = {};
      c = WMMA_BF16(Ah[0], *(const v16bf*)&sB3[j][lane][0], c);
      c = WMMA_BF16(Ah[1], *(const v16bf*)&sB3[3 + j][lane][0], c);
      const int col = 16 * j + mrow;
      if (col < 8) {
#pragma unroll
        for (int i = 0; i < 8; ++i)
          sLogit[sbase + i + 8 * hi][col] = 0.25f * c[i];   // / sqrt(DH)
      } else if (col < 40) {
#pragma unroll
        for (int i = 0; i < 8; ++i)
          sCoeff[sbase + i + 8 * hi][col - 8] = c[i];
      }
    }
    __syncthreads();
  }

  // ---- segment softmax over senders: wave w owns head w ----
  {
    const int h = w;
    float lg[8], mx = -1e30f;
#pragma unroll
    for (int i = 0; i < 8; ++i) {
      const int s = lane + 32 * i;
      lg[i] = sLogit[s][h];
      if (s != r) mx = fmaxf(mx, lg[i]);
    }
#pragma unroll
    for (int off = 16; off > 0; off >>= 1) mx = fmaxf(mx, __shfl_xor(mx, off));
    float ex[8], sum = 0.f;
#pragma unroll
    for (int i = 0; i < 8; ++i) {
      const int s = lane + 32 * i;
      ex[i] = (s == r) ? 0.f : __expf(lg[i] - mx);
      sum += ex[i];
    }
#pragma unroll
    for (int off = 16; off > 0; off >>= 1) sum += __shfl_xor(sum, off);
    const float rs = __builtin_amdgcn_rcpf(sum + 1e-12f);
#pragma unroll
    for (int i = 0; i < 8; ++i) sLogit[lane + 32 * i][h] = ex[i] * rs;
  }
  __syncthreads();

  // ---- aggregation: wave w handles senders [32w, 32w+32); lane l owns
  //      inv channels 4l..4l+3 and vector channel v=l (same head l>>2) ----
  {
    float ia0 = 0, ia1 = 0, ia2 = 0, ia3 = 0, va0 = 0, va1 = 0, va2 = 0;
    const int hd = lane >> 2;
    const float prx = sPosR[lane * 3 + 0];
    const float pry = sPosR[lane * 3 + 1];
    const float prz = sPosR[lane * 3 + 2];
    for (int ii = 0; ii < 32; ++ii) {
      const int s = w * 32 + ii;
      if (s == r) continue;
      const float a = sLogit[s][hd];
      const float4 vp = *(const float4*)(wsV + s * 128 + 4 * lane);
      ia0 = fmaf(a, vp.x, ia0); ia1 = fmaf(a, vp.y, ia1);
      ia2 = fmaf(a, vp.z, ia2); ia3 = fmaf(a, vp.w, ia3);
      const float dx = pos[(s * 32 + lane) * 3 + 0] - prx;
      const float dy = pos[(s * 32 + lane) * 3 + 1] - pry;
      const float dz = pos[(s * 32 + lane) * 3 + 2] - prz;
      const float rinv =
          __builtin_amdgcn_rsqf(fmaf(dx, dx, fmaf(dy, dy, fmaf(dz, dz, 1e-18f))));
      const float wt = a * sCoeff[s][lane] * 1.7320508075688772f * rinv;
      va0 = fmaf(wt, dx, va0); va1 = fmaf(wt, dy, va1); va2 = fmaf(wt, dz, va2);
    }
    sPInv[w][4 * lane + 0] = ia0; sPInv[w][4 * lane + 1] = ia1;
    sPInv[w][4 * lane + 2] = ia2; sPInv[w][4 * lane + 3] = ia3;
    sPVec[w][3 * lane + 0] = va0; sPVec[w][3 * lane + 1] = va1;
    sPVec[w][3 * lane + 2] = va2;
  }
  __syncthreads();
  if (tid < 128) {            // deterministic ordered reduction (no atomics)
    float a = 0.f;
#pragma unroll
    for (int p = 0; p < 8; ++p) a += sPInv[p][tid];
    wsInv[r * 128 + tid] = a;
  }
  if (tid < 96) {
    float a = 0.f;
#pragma unroll
    for (int p = 0; p < 8; ++p) a += sPVec[p][tid];
    wsVec[r * 96 + tid] = a;
  }
}

// ---------------------------------------------------------------------------
// Kernel 3: per-node finalize: vec mix + residual, node MLP + residual
// ---------------------------------------------------------------------------
__global__ __launch_bounds__(128) void finalize_kernel(
    const float* __restrict__ pos, const float* __restrict__ feat,
    const float* __restrict__ W_mix,
    const float* __restrict__ M1, const float* __restrict__ bm1,
    const float* __restrict__ M2, const float* __restrict__ bm2,
    const float* __restrict__ M3, const float* __restrict__ bm3,
    const float* __restrict__ wsInv, const float* __restrict__ wsVec,
    float* __restrict__ out) {
  __shared__ float sInv[128], sH1[64], sH2[64], sVg[96];
  const int n = blockIdx.x, t = threadIdx.x;
  sInv[t] = wsInv[n * 128 + t];
  if (t < 96) sVg[t] = wsVec[n * 96 + t];
  __syncthreads();
  if (t < 96) {
    const int wv = t / 3, c = t - 3 * wv;
    float a = 0.f;
#pragma unroll 8
    for (int v = 0; v < 32; ++v) a = fmaf(sVg[v * 3 + c], W_mix[v * 32 + wv], a);
    out[n * 96 + t] = a + pos[n * 96 + t];
  }
  if (t < 64) {
    float a = bm1[t];
#pragma unroll 8
    for (int d = 0; d < 128; ++d) a = fmaf(sInv[d], M1[d * 64 + t], a);
    sH1[t] = selu_f(a);
  }
  __syncthreads();
  if (t < 64) {
    float a = bm2[t];
#pragma unroll 8
    for (int d = 0; d < 64; ++d) a = fmaf(sH1[d], M2[d * 64 + t], a);
    sH2[t] = selu_f(a);
  }
  __syncthreads();
  {
    float a = bm3[t];
#pragma unroll 8
    for (int d = 0; d < 64; ++d) a = fmaf(sH2[d], M3[d * 128 + t], a);
    out[NN * 96 + n * 128 + t] = a + feat[n * 128 + t];
  }
}

// ---------------------------------------------------------------------------
extern "C" void kernel_launch(void* const* d_in, const int* in_sizes, int n_in,
                              void* d_out, int out_size, void* d_ws, size_t ws_size,
                              hipStream_t stream) {
  (void)in_sizes; (void)n_in; (void)out_size; (void)ws_size;
  const float* pos   = (const float*)d_in[0];
  const float* feat  = (const float*)d_in[1];
  const float* We1   = (const float*)d_in[2];
  const float* be1   = (const float*)d_in[3];
  const float* We2   = (const float*)d_in[4];
  const float* be2   = (const float*)d_in[5];
  const float* W_q   = (const float*)d_in[6];
  const float* W_k   = (const float*)d_in[7];
  const float* W_v   = (const float*)d_in[8];
  const float* W_vec = (const float*)d_in[9];
  const float* W_mix = (const float*)d_in[10];
  const float* M1    = (const float*)d_in[11];
  const float* bm1   = (const float*)d_in[12];
  const float* M2    = (const float*)d_in[13];
  const float* bm2   = (const float*)d_in[14];
  const float* M3    = (const float*)d_in[15];
  const float* bm3   = (const float*)d_in[16];

  float*  ws     = (float*)d_ws;
  float*  wsFs   = ws;                 // [256*64]
  float*  wsFr   = ws + 16384;         // [256*64]
  float*  wsQ    = ws + 32768;         // [256*128]
  float*  wsV    = ws + 65536;         // [256*128]
  float*  wsInv  = ws + 98304;         // [256*128]
  float*  wsVec  = ws + 131072;        // [256*96]
  __bf16* wsPack = (__bf16*)(ws + 155648); // [40*512] bf16 = 40KB

  pack_weights_kernel<<<1, 256, 0, stream>>>(We1, We2, wsPack);
  node_pre_kernel<<<NN, 128, 0, stream>>>(feat, We1, be1, W_q, W_v,
                                          wsFs, wsFr, wsQ, wsV);
  edge_attn_kernel<<<NN, 256, 0, stream>>>(pos, be2, W_k, W_vec, wsPack,
                                           wsFs, wsFr, wsQ, wsV, wsInv, wsVec);
  finalize_kernel<<<NN, 128, 0, stream>>>(pos, feat, W_mix, M1, bm1, M2, bm2,
                                          M3, bm3, wsInv, wsVec, (float*)d_out);
}
